// RoFormerSelfAttention_71554155152341
// MI455X (gfx1250) — compile-verified, matrix operand order checked by reading, not run on hardware
//
#include <hip/hip_runtime.h>

typedef _Float16 v16h __attribute__((ext_vector_type(16)));
typedef _Float16 v4h  __attribute__((ext_vector_type(4)));
typedef float    v8f  __attribute__((ext_vector_type(8)));

// Problem constants
#define BB 4
#define SS 2048
#define HDIM 1024
#define NH 16
#define DD 64

// 16-bit A/B fragment K-index mapping (ISA 7.12.2): element e, half = (lane>>4)&1
__device__ __forceinline__ int kmap8(int e, int hi) { return e + (e & 8) + (hi << 3); }

// ---------------------------------------------------------------------------
// Async DMA global -> LDS (ASYNCcnt-tracked), via inline asm so we do not
// depend on toolchain-specific builtin signatures (bridge-doc recommendation).
// GV mode: VDST = LDS byte address VGPR, VADDR = 64-bit global address pair.
// Low 32 bits of a generic LDS pointer are the LDS byte offset (aperture
// check uses addr[63:32] only).
// ---------------------------------------------------------------------------
__device__ __forceinline__ void async_cp16(const _Float16* g, _Float16* l) {
    unsigned int lds_off = (unsigned int)(unsigned long long)(void*)l;
    unsigned long long gaddr = (unsigned long long)(const void*)g;
    asm volatile("global_load_async_to_lds_b128 %0, %1, off"
                 :
                 : "v"(lds_off), "v"(gaddr)
                 : "memory");
}

__device__ __forceinline__ void wait_async0() {
#if __has_builtin(__builtin_amdgcn_s_wait_asynccnt)
    __builtin_amdgcn_s_wait_asynccnt(0);
#else
    asm volatile("s_wait_asynccnt 0" ::: "memory");
#endif
}

// ---------------------------------------------------------------------------
// Kernel 1: fused QKV projection (out = hs @ W^T + b) + interleaved rotary on
// Q and K. grid = (M/64, N/64, 3). block = 256 (8 waves). Output f16 [B,NH,S,D].
// ---------------------------------------------------------------------------
__global__ __launch_bounds__(256) void qkv_rope_kernel(
    const float* __restrict__ hs, const float* __restrict__ rel,
    const float* __restrict__ Wq, const float* __restrict__ bq,
    const float* __restrict__ Wk, const float* __restrict__ bk,
    const float* __restrict__ Wv, const float* __restrict__ bv,
    _Float16* __restrict__ out)
{
    __shared__ _Float16 As[64][40];   // 64 rows x 32 K, padded
    __shared__ _Float16 Bs[64][40];

    const int tid  = threadIdx.x;
    const int lane = tid & 31;
    const int wid  = tid >> 5;
    const int hi   = (lane >> 4) & 1;
    const int ln15 = lane & 15;
    const int wm   = wid & 3;         // wave tile row (0..3)
    const int wn   = wid >> 2;        // wave tile col (0..1), 32 cols each
    const int m0   = blockIdx.x * 64;
    const int n0   = blockIdx.y * 64;
    const int z    = blockIdx.z;

    const float* W    = (z == 0) ? Wq : (z == 1 ? Wk : Wv);
    const float* bias = (z == 0) ? bq : (z == 1 ? bk : bv);
    _Float16*    dst  = out + (size_t)z * ((size_t)BB * NH * SS * DD);

    v8f acc[2];
    for (int j = 0; j < 2; ++j)
        for (int r = 0; r < 8; ++r) acc[j][r] = 0.f;

    for (int kt = 0; kt < HDIM; kt += 32) {
        // Fill LDS: float4 global loads -> packed 4xf16 (ds_store_b64)
        for (int i = 0; i < 2; ++i) {
            int idx = i * 256 + tid;       // 512 chunks of 4 elements
            int r = idx >> 3;              // 8 chunks per 32-wide row
            int c = (idx & 7) << 2;
            const float4 xa = *(const float4*)(hs + (size_t)(m0 + r) * HDIM + kt + c);
            const float4 xb = *(const float4*)(W  + (size_t)(n0 + r) * HDIM + kt + c);
            v4h pa, pb;
            pa[0] = (_Float16)xa.x; pa[1] = (_Float16)xa.y;
            pa[2] = (_Float16)xa.z; pa[3] = (_Float16)xa.w;
            pb[0] = (_Float16)xb.x; pb[1] = (_Float16)xb.y;
            pb[2] = (_Float16)xb.z; pb[3] = (_Float16)xb.w;
            *(v4h*)&As[r][c] = pa;
            *(v4h*)&Bs[r][c] = pb;
        }
        __syncthreads();

        v16h a;
        for (int e = 0; e < 16; ++e) a[e] = As[wm * 16 + ln15][kmap8(e, hi)];
        for (int j = 0; j < 2; ++j) {
            v16h bf;
            for (int e = 0; e < 16; ++e)
                bf[e] = Bs[wn * 32 + j * 16 + ln15][kmap8(e, hi)];
            acc[j] = __builtin_amdgcn_wmma_f32_16x16x32_f16(
                false, a, false, bf, (short)0, acc[j], false, false);
        }
        __syncthreads();
    }

    // Epilogue: bias + interleaved rotary (pair (d, d^1) == (lane, lane^1))
    for (int j = 0; j < 2; ++j) {
        const int n = n0 + wn * 32 + j * 16 + ln15;
        const int d = n & 63, h = n >> 6;
        for (int r = 0; r < 8; ++r) {
            const int m  = m0 + wm * 16 + r + 8 * hi;
            const int bi = m >> 11, s = m & (SS - 1);
            float v = acc[j][r] + bias[n];
            if (z < 2) {
                const float cv = rel[s * DD + ((d >> 1) << 1) + 1]; // odd slots -> cos
                const float sv = rel[s * DD + ((d >> 1) << 1)];     // even slots -> sin
                const float vp = __shfl_xor(v, 1, 32);              // partner value x[d^1]
                const float rot = (d & 1) ? vp : -vp;               // even d: -x[d+1]; odd d: x[d-1]
                v = v * cv + rot * sv;
            }
            dst[((size_t)(bi * NH + h) * SS + s) * DD + d] = (_Float16)v;
        }
    }
}

// ---------------------------------------------------------------------------
// Kernel 2: flash attention. grid = (S/64, B*NH). block = 128 (4 waves).
// Each wave owns 16 query rows; loop over 64-key blocks with online softmax.
// ---------------------------------------------------------------------------
__global__ __launch_bounds__(128) void flash_attn_kernel(
    const _Float16* __restrict__ qb, const _Float16* __restrict__ kbuf,
    const _Float16* __restrict__ vbuf, const float* __restrict__ mask,
    float* __restrict__ out)
{
    __shared__ _Float16 Ks[64][72];
    __shared__ _Float16 Vs[64][72];
    __shared__ _Float16 Ps[4][16][72];

    const int tid  = threadIdx.x;
    const int lane = tid & 31;
    const int wave = tid >> 5;
    const int hi   = (lane >> 4) & 1;
    const int ln15 = lane & 15;
    const int qblk = blockIdx.x;           // query block (64 rows)
    const int bh   = blockIdx.y;
    const int b    = bh >> 4, h = bh & 15;

    const size_t headoff = (size_t)bh * SS * DD;
    const _Float16* qh = qb   + headoff;
    const _Float16* kh = kbuf + headoff;
    const _Float16* vh = vbuf + headoff;

    // Q A-fragments, kept in registers for the whole kernel (D=64 -> 2 chunks)
    v16h aq[2];
    {
        const int srow = qblk * 64 + wave * 16 + ln15;
        for (int kc = 0; kc < 2; ++kc)
            for (int e = 0; e < 16; ++e)
                aq[kc][e] = qh[(size_t)srow * DD + kc * 32 + kmap8(e, hi)];
    }

    v8f o[4];
    float m_s[8], l_s[8];
    for (int j = 0; j < 4; ++j)
        for (int r = 0; r < 8; ++r) o[j][r] = 0.f;
    for (int r = 0; r < 8; ++r) { m_s[r] = -1e30f; l_s[r] = 0.f; }

    const int row  = tid >> 1;            // 0..63
    const int hoff = (tid & 1) * 32;      // half-row offset (32 halves = 64B)

    for (int kb = 0; kb < SS / 64; ++kb) {
        // Cooperative 64x64 f16 tile loads into LDS via async DMA (ASYNCcnt)
        {
            const _Float16* sk = kh + (size_t)(kb * 64 + row) * DD + hoff;
            const _Float16* sv = vh + (size_t)(kb * 64 + row) * DD + hoff;
            for (int i = 0; i < 4; ++i) {
                async_cp16(sk + i * 8, &Ks[row][hoff + i * 8]);
                async_cp16(sv + i * 8, &Vs[row][hoff + i * 8]);
            }
            wait_async0();
        }
        if (kb + 1 < SS / 64) {   // global_prefetch_b8 for next block
            __builtin_prefetch(kh + (size_t)((kb + 1) * 64 + row) * DD + hoff, 0, 0);
            __builtin_prefetch(vh + (size_t)((kb + 1) * 64 + row) * DD + hoff, 0, 0);
        }
        __syncthreads();

        // scores: 4 N-tiles x 2 K-chunks of WMMA
        v8f st[4];
        for (int j = 0; j < 4; ++j) {
            for (int r = 0; r < 8; ++r) st[j][r] = 0.f;
            for (int kc = 0; kc < 2; ++kc) {
                v16h bk2;
                for (int e = 0; e < 16; ++e)
                    bk2[e] = Ks[j * 16 + ln15][kc * 32 + kmap8(e, hi)];
                st[j] = __builtin_amdgcn_wmma_f32_16x16x32_f16(
                    false, aq[kc], false, bk2, (short)0, st[j], false, false);
            }
            const float mk = mask[(size_t)b * SS + kb * 64 + j * 16 + ln15];
            for (int r = 0; r < 8; ++r) st[j][r] = st[j][r] * 0.125f + mk;
        }

        // online softmax; row stats live across each half-wave (width-16 shfl)
        float p[4][8];
        for (int r = 0; r < 8; ++r) {
            float mx = st[0][r];
            for (int j = 1; j < 4; ++j) mx = fmaxf(mx, st[j][r]);
            for (int off = 1; off < 16; off <<= 1) mx = fmaxf(mx, __shfl_xor(mx, off, 16));
            const float nm    = fmaxf(m_s[r], mx);
            const float alpha = __expf(m_s[r] - nm);
            float rs = 0.f;
            for (int j = 0; j < 4; ++j) { p[j][r] = __expf(st[j][r] - nm); rs += p[j][r]; }
            for (int off = 1; off < 16; off <<= 1) rs += __shfl_xor(rs, off, 16);
            l_s[r] = l_s[r] * alpha + rs;
            m_s[r] = nm;
            for (int j = 0; j < 4; ++j) o[j][r] *= alpha;
        }

        // stage P (C-layout) -> LDS so it can be re-read in A-fragment layout
        for (int j = 0; j < 4; ++j)
            for (int r = 0; r < 8; ++r)
                Ps[wave][r + 8 * hi][j * 16 + ln15] = (_Float16)p[j][r];
        __syncthreads();

        // PV: o += P (16x64) * V (64x64)
        for (int kc = 0; kc < 2; ++kc) {
            v16h ap;
            for (int e = 0; e < 16; ++e)
                ap[e] = Ps[wave][ln15][kc * 32 + kmap8(e, hi)];
            for (int j = 0; j < 4; ++j) {
                v16h bv2;
                for (int e = 0; e < 16; ++e)
                    bv2[e] = Vs[kc * 32 + kmap8(e, hi)][j * 16 + ln15];
                o[j] = __builtin_amdgcn_wmma_f32_16x16x32_f16(
                    false, ap, false, bv2, (short)0, o[j], false, false);
            }
        }
        __syncthreads();
    }

    // epilogue: ctx -> [B, S, NH*D] fp32
    for (int j = 0; j < 4; ++j) {
        const int d = j * 16 + ln15;
        for (int r = 0; r < 8; ++r) {
            const int s = qblk * 64 + wave * 16 + r + 8 * hi;
            out[((size_t)(b * SS + s) * HDIM) + h * DD + d] = o[j][r] / l_s[r];
        }
    }
}

// ---------------------------------------------------------------------------
extern "C" void kernel_launch(void* const* d_in, const int* in_sizes, int n_in,
                              void* d_out, int out_size, void* d_ws, size_t ws_size,
                              hipStream_t stream) {
    const float* hs   = (const float*)d_in[0];
    const float* mask = (const float*)d_in[1];
    const float* rel  = (const float*)d_in[2];
    const float* Wq   = (const float*)d_in[3];
    const float* bq   = (const float*)d_in[4];
    const float* Wk   = (const float*)d_in[5];
    const float* bk   = (const float*)d_in[6];
    const float* Wv   = (const float*)d_in[7];
    const float* bv   = (const float*)d_in[8];
    float* out = (float*)d_out;

    _Float16* ws = (_Float16*)d_ws;
    const size_t per = (size_t)BB * NH * SS * DD;   // 8,388,608 halves (16 MB)
    _Float16* qb = ws;
    _Float16* kb = ws + per;
    _Float16* vb = ws + 2 * per;

    dim3 g1((BB * SS) / 64, HDIM / 64, 3), b1(256);
    qkv_rope_kernel<<<g1, b1, 0, stream>>>(hs, rel, Wq, bq, Wk, bk, Wv, bv, ws);

    dim3 g2(SS / 64, BB * NH), b2(128);
    flash_attn_kernel<<<g2, b2, 0, stream>>>(qb, kb, vb, mask, out);
}
